// MyCorrelation_12137577579279
// MI455X (gfx1250) — matrix operand order repaired and checked by reading, not compiled, not run on hardware
//
#include <hip/hip_runtime.h>

// CDNA5 / gfx1250 WMMA correlation (cost volume) kernel, v2.
// out[b,(di,dj),h,w] = (1/64) * sum_c feat1[b,c,h,w] * feat2[b,c,h+di,w+dj]
//
// Per 16-pixel strip the 9 dj-diagonals are the band of GEMM A(16xC)*B(Cx24)^T;
// covered by two 16-wide N tiles (w0-8, w0+8) with v_wmma_f32_16x16x32_bf16
// (K=64 -> 2 chained WMMAs per N tile). Diagonals are extracted via an LDS
// bounce so output stores are contiguous 64B rows.
//
// fp32->bf16 via single v_perm_b32 per packed pair (truncation; bf16 keeps the
// full fp32 exponent range, fp32 accumulation in the WMMA).

typedef __bf16        v16bf __attribute__((ext_vector_type(16)));
typedef float         v8f   __attribute__((ext_vector_type(8)));
typedef unsigned int  v8u   __attribute__((ext_vector_type(8)));

#define DMAX 4
#define ND   9
#define NDISP 81
#define Bn 4
#define Cc 64
#define Hh 192
#define Ww 448
#define TILES_W (Ww / 16)        // 28
#define WAVES_PER_BLOCK 4
#define LDS_STRIDE 33            // 32 cols + 1 pad (bank-conflict-free)

static __device__ __forceinline__ unsigned pack_bf16x2(float lo, float hi) {
    // dst = { hi[31:16], lo[31:16] } : one v_perm_b32
    return __builtin_amdgcn_perm(__builtin_bit_cast(unsigned, hi),
                                 __builtin_bit_cast(unsigned, lo),
                                 0x07060302u);
}

// Build the four B fragments (two N tiles x two K chunks) for one feat2 row.
// GUARD=false: interior tile, no column checks (wave-uniform fast path).
template <bool GUARD>
static __device__ __forceinline__ void load_b_frags(
        const float* __restrict__ f2row, int HW,
        int w2a, int w2b, bool colA, bool colB, int hiHalf,
        v16bf& b00, v16bf& b01, v16bf& b10, v16bf& b11) {
    v8u u00, u01, u10, u11;
    #pragma unroll
    for (int r = 0; r < 8; ++r) {
        const int k = ((r < 4) ? 2 * r : 2 * r + 8) + hiHalf * 8;
        const float* p0 = f2row + (size_t)k * HW;          // K = k
        const float* p1 = f2row + (size_t)(k + 1) * HW;    // K = k+1
        const float* p2 = f2row + (size_t)(k + 32) * HW;   // K chunk 1
        const float* p3 = f2row + (size_t)(k + 33) * HW;
        float aL, aH, a2L, a2H, bL, bH, b2L, b2H;
        if (!GUARD || colA) { aL = p0[w2a]; aH = p1[w2a]; a2L = p2[w2a]; a2H = p3[w2a]; }
        else                { aL = aH = a2L = a2H = 0.0f; }
        if (!GUARD || colB) { bL = p0[w2b]; bH = p1[w2b]; b2L = p2[w2b]; b2H = p3[w2b]; }
        else                { bL = bH = b2L = b2H = 0.0f; }
        u00[r] = pack_bf16x2(aL,  aH);
        u01[r] = pack_bf16x2(a2L, a2H);
        u10[r] = pack_bf16x2(bL,  bH);
        u11[r] = pack_bf16x2(b2L, b2H);
    }
    b00 = __builtin_bit_cast(v16bf, u00);
    b01 = __builtin_bit_cast(v16bf, u01);
    b10 = __builtin_bit_cast(v16bf, u10);
    b11 = __builtin_bit_cast(v16bf, u11);
}

__global__ __launch_bounds__(128) void corr81_wmma_bf16(
        const float* __restrict__ f1,
        const float* __restrict__ f2,
        float* __restrict__ out) {
    __shared__ float ldsbuf[WAVES_PER_BLOCK * 16 * LDS_STRIDE];

    const int HW   = Hh * Ww;
    const int lane = threadIdx.x & 31;
    const int waveInBlock = threadIdx.x >> 5;
    const int waveId = blockIdx.x * WAVES_PER_BLOCK + waveInBlock;
    float* myLds = ldsbuf + waveInBlock * (16 * LDS_STRIDE);

    // waveId -> (b, h, tile)
    const int tw = waveId % TILES_W;
    const int t1 = waveId / TILES_W;
    const int h  = t1 % Hh;
    const int b  = t1 / Hh;
    const int w0 = tw * 16;

    const int m      = lane & 15;    // A: M row | B/C/D: N col
    const int hiHalf = lane >> 4;

    // ---- A fragments: feat1[b, :, h, w0+m]; packed-pair K mapping ----------
    // elem pair (2r,2r+1): K = (r<4 ? 2r : 2r+8) + hiHalf*8 (and +32 chunk)
    v16bf a0, a1;
    {
        const float* f1p = f1 + (size_t)b * Cc * HW + (size_t)h * Ww + (w0 + m);
        v8u ua0, ua1;
        #pragma unroll
        for (int r = 0; r < 8; ++r) {
            const int k = ((r < 4) ? 2 * r : 2 * r + 8) + hiHalf * 8;
            ua0[r] = pack_bf16x2(f1p[(size_t)k * HW],        f1p[(size_t)(k + 1) * HW]);
            ua1[r] = pack_bf16x2(f1p[(size_t)(k + 32) * HW], f1p[(size_t)(k + 33) * HW]);
        }
        a0 = __builtin_bit_cast(v16bf, ua0);
        a1 = __builtin_bit_cast(v16bf, ua1);
    }

    const int  w2a  = w0 - 8 + m;                       // N tile 0 column
    const int  w2b  = w0 + 8 + m;                       // N tile 1 column
    const bool colA = ((unsigned)w2a < (unsigned)Ww);
    const bool colB = ((unsigned)w2b < (unsigned)Ww);
    const bool interior = (tw != 0) && (tw != TILES_W - 1);   // wave-uniform

    const float inv_c = 1.0f / (float)Cc;
    const float* f2base = f2 + (size_t)b * Cc * HW;

    for (int di = -DMAX; di <= DMAX; ++di) {
        const int h2 = h + di;
        const bool hok = ((unsigned)h2 < (unsigned)Hh);       // wave-uniform
        const size_t outBase =
            (((size_t)b * NDISP + (size_t)(di + DMAX) * ND) * Hh + h) * Ww + w0;

        if (!hok) {
            // entire di-slab is zero (zero padding); store 9 x 16 zeros
            #pragma unroll
            for (int it = 0; it < 5; ++it) {
                const int idx = lane + it * 32;           // idx = dj_i*16 + M
                if (idx < ND * 16) {
                    const int dj_i = idx >> 4;
                    const int M    = idx & 15;
                    out[outBase + (size_t)dj_i * (Hh * Ww) + M] = 0.0f;
                }
            }
            continue;
        }

        const float* f2row = f2base + (size_t)h2 * Ww;

        // prefetch next displacement's row (global_prefetch_b8)
        if (di < DMAX && (unsigned)(h2 + 1) < (unsigned)Hh)
            __builtin_prefetch(f2base + (size_t)(h2 + 1) * Ww + w0, 0, 1);

        v16bf b00, b01, b10, b11;
        if (interior)
            load_b_frags<false>(f2row, HW, w2a, w2b, colA, colB, hiHalf,
                                b00, b01, b10, b11);
        else
            load_b_frags<true>(f2row, HW, w2a, w2b, colA, colB, hiHalf,
                               b00, b01, b10, b11);

        // ---- 16x16x64 = two chained 16x16x32 bf16 WMMAs per N tile --------
        v8f c0 = {};
        v8f c1 = {};
        c0 = __builtin_amdgcn_wmma_f32_16x16x32_bf16(false, a0, false, b00,
                                                     (short)0, c0, false, false);
        c0 = __builtin_amdgcn_wmma_f32_16x16x32_bf16(false, a1, false, b01,
                                                     (short)0, c0, false, false);
        c1 = __builtin_amdgcn_wmma_f32_16x16x32_bf16(false, a0, false, b10,
                                                     (short)0, c1, false, false);
        c1 = __builtin_amdgcn_wmma_f32_16x16x32_bf16(false, a1, false, b11,
                                                     (short)0, c1, false, false);

        // ---- LDS bounce: C layout lane l -> N=l&15, VGPR r -> M=r+(l>>4)*8
        #pragma unroll
        for (int r = 0; r < 8; ++r) {
            const int M = r + hiHalf * 8;
            myLds[M * LDS_STRIDE + m]      = c0[r];   // N tile 0: cols 0..15
            myLds[M * LDS_STRIDE + m + 16] = c1[r];   // N tile 1: cols 16..31
        }
        // (same-wave DS ops are in-order; compiler inserts s_wait_dscnt)

        // ---- read 9 band diagonals, store contiguous 64B output rows ------
        // value(M, dj) at LDS col = M + dj + 8 ; dj = dj_i - 4 -> col=M+dj_i+4
        #pragma unroll
        for (int it = 0; it < 5; ++it) {
            const int idx = lane + it * 32;               // dj_i*16 + M
            if (idx < ND * 16) {
                const int dj_i = idx >> 4;
                const int M    = idx & 15;
                const float v  = myLds[M * LDS_STRIDE + (M + dj_i + 4)] * inv_c;
                out[outBase + (size_t)dj_i * (Hh * Ww) + M] = v;
            }
        }
    }
}

extern "C" void kernel_launch(void* const* d_in, const int* in_sizes, int n_in,
                              void* d_out, int out_size, void* d_ws, size_t ws_size,
                              hipStream_t stream) {
    (void)in_sizes; (void)n_in; (void)out_size; (void)d_ws; (void)ws_size;
    const float* f1 = (const float*)d_in[0];
    const float* f2 = (const float*)d_in[1];
    float* out = (float*)d_out;

    const int totalWaves = Bn * Hh * TILES_W;            // 21504
    const int blocks = totalWaves / WAVES_PER_BLOCK;     // 5376 (exact)
    hipLaunchKernelGGL(corr81_wmma_bf16, dim3(blocks), dim3(128), 0, stream,
                       f1, f2, out);
}
